// GlobalSSM_21157008900533
// MI455X (gfx1250) — matrix operand order repaired and checked by reading, not compiled, not run on hardware
//
#include <hip/hip_runtime.h>
#include <hip/hip_bf16.h>

typedef __attribute__((ext_vector_type(16))) __bf16 v16bf;
typedef __attribute__((ext_vector_type(8)))  float  v8f;
typedef __attribute__((ext_vector_type(2)))  float  v2f;

#define WMMA_BF16(a,b,c) __builtin_amdgcn_wmma_f32_16x16x32_bf16(false,(a),false,(b),(short)0,(c),false,false)
#define WMMA_F32(a,b,c)  __builtin_amdgcn_wmma_f32_16x16x4_f32 (false,(a),false,(b),(short)0,(c),false,false)

// Problem constants
#define BB 8
#define SS 4096
#define DD 1024
#define NN 128
#define CHUNKS 64
#define CLEN 64
#define HSTR 130   // LDS row stride (floats), avoids 128-stride bank conflicts

// ---------------------------------------------------------------------------
// Pack B_mat (D x N, f32 row-major) into bf16 WMMA B-operand fragment order:
// [kb(32)][nb(8)][lane(32)][e(16)], lane<16: col=nb*16+lane, K=kb*32+e
//                            lane>=16: col=nb*16+(lane-16), K=kb*32+16+e
// ---------------------------------------------------------------------------
__global__ void k_packB(const float* __restrict__ Bm, __bf16* __restrict__ pk) {
  int idx = blockIdx.x * 256 + threadIdx.x;        // 131072 total
  int e  = idx & 15;
  int l  = (idx >> 4) & 31;
  int nb = (idx >> 9) & 7;
  int kb = idx >> 12;
  int k  = kb * 32 + ((l < 16) ? 0 : 16) + e;
  int n  = nb * 16 + (l & 15);
  pk[idx] = (__bf16)Bm[k * NN + n];
}

// ---------------------------------------------------------------------------
// u = x @ B_mat : [32768,1024] x [1024,128] -> f32 u[32768,128]
// One wave per 16 M-rows; 8 N-tiles in registers; K in 32-element chunks.
// Packed-B slices are staged in LDS with async global->LDS copies (ASYNCcnt),
// double-buffered so slice kb+1 streams in while slice kb feeds the WMMAs.
// ---------------------------------------------------------------------------
__global__ void k_gemm_u(const float* __restrict__ x,
                         const __bf16* __restrict__ pk,
                         float* __restrict__ u) {
  __shared__ __align__(16) __bf16 spk[2][4096];    // 2 x 8KB slices
  int tid  = threadIdx.x;
  int l    = tid & 31;
  int wave = blockIdx.x * (blockDim.x >> 5) + (tid >> 5);
  int m0   = wave * 16;
  int ln   = l & 15;
  int hi   = l >> 4;                               // 0 or 1
  const float* xrow = x + (size_t)(m0 + ln) * DD;

  // LDS byte offsets (low 32 bits of a generic pointer to LDS == LDS offset)
  unsigned lds_base = (unsigned)(size_t)(&spk[0][0]);
  const char* pkb = (const char*)pk;

  // Prologue: stage slice kb=0 into buffer 0 (each thread copies 2 x 16B)
#pragma unroll
  for (int j = 0; j < 2; ++j) {
    unsigned loff = lds_base + (unsigned)(tid * 16 + j * 4096);
    unsigned long long g = (unsigned long long)(size_t)(pkb + tid * 16 + j * 4096);
    asm volatile("global_load_async_to_lds_b128 %0, %1, off"
                 :: "v"(loff), "v"(g) : "memory");
  }

  v8f acc[8];
  v8f z = {0.f,0.f,0.f,0.f,0.f,0.f,0.f,0.f};
#pragma unroll
  for (int nb = 0; nb < 8; ++nb) acc[nb] = z;

  int buf = 0;
  for (int kb = 0; kb < 32; ++kb) {
    // Issue x-tile loads early so they overlap the slice handoff barriers.
    int k0 = kb * 32 + hi * 16;
    if (kb < 30) __builtin_prefetch(xrow + k0 + 64, 0, 3);
    float4 a0 = *(const float4*)(xrow + k0);
    float4 a1 = *(const float4*)(xrow + k0 + 4);
    float4 a2 = *(const float4*)(xrow + k0 + 8);
    float4 a3 = *(const float4*)(xrow + k0 + 12);

    __syncthreads();   // everyone done reading buf^1 -> safe to overwrite
    if (kb + 1 < 32) {
#pragma unroll
      for (int j = 0; j < 2; ++j) {
        unsigned loff = lds_base + (unsigned)((buf ^ 1) * 8192 + tid * 16 + j * 4096);
        unsigned long long g =
            (unsigned long long)(size_t)(pkb + (size_t)(kb + 1) * 8192 + tid * 16 + j * 4096);
        asm volatile("global_load_async_to_lds_b128 %0, %1, off"
                     :: "v"(loff), "v"(g) : "memory");
      }
      asm volatile("s_wait_asynccnt 0x2" ::: "memory");  // slice kb landed
    } else {
      asm volatile("s_wait_asynccnt 0x0" ::: "memory");
    }
    __syncthreads();   // slice kb visible to all waves

    v16bf af;
    af[0]=(__bf16)a0.x; af[1]=(__bf16)a0.y; af[2]=(__bf16)a0.z; af[3]=(__bf16)a0.w;
    af[4]=(__bf16)a1.x; af[5]=(__bf16)a1.y; af[6]=(__bf16)a1.z; af[7]=(__bf16)a1.w;
    af[8]=(__bf16)a2.x; af[9]=(__bf16)a2.y; af[10]=(__bf16)a2.z; af[11]=(__bf16)a2.w;
    af[12]=(__bf16)a3.x; af[13]=(__bf16)a3.y; af[14]=(__bf16)a3.z; af[15]=(__bf16)a3.w;

    // Batch all 8 fragment reads (ds_load_b128 pairs), then the 8 WMMAs.
    v16bf bfr[8];
#pragma unroll
    for (int nb = 0; nb < 8; ++nb)
      bfr[nb] = *(const v16bf*)(&spk[buf][(nb * 32 + l) * 16]);
#pragma unroll
    for (int nb = 0; nb < 8; ++nb)
      acc[nb] = WMMA_BF16(af, bfr[nb], acc[nb]);

    buf ^= 1;
  }

#pragma unroll
  for (int nb = 0; nb < 8; ++nb)
#pragma unroll
    for (int r = 0; r < 8; ++r)
      u[(size_t)(m0 + hi * 8 + r) * NN + nb * 16 + ln] = acc[nb][r];
}

// ---------------------------------------------------------------------------
// Q = P @ P  (128x128 f32) using v_wmma_f32_16x16x4_f32. 64 waves, 1 tile each.
// ---------------------------------------------------------------------------
__global__ void k_matsq(const float* __restrict__ P, float* __restrict__ Q) {
  int l  = threadIdx.x & 31;
  int w  = blockIdx.x * (blockDim.x >> 5) + (threadIdx.x >> 5);  // 0..63
  int mt = w >> 3, nt = w & 7;
  int ln = l & 15;
  int hi = l >> 4;
  v8f c = {0.f,0.f,0.f,0.f,0.f,0.f,0.f,0.f};
  for (int kb = 0; kb < 32; ++kb) {
    int ka = kb * 4 + hi * 2;
    v2f a, b;
    a.x = P[(mt * 16 + ln) * NN + ka];
    a.y = P[(mt * 16 + ln) * NN + ka + 1];
    b.x = P[(size_t)ka * NN + nt * 16 + ln];
    b.y = P[(size_t)(ka + 1) * NN + nt * 16 + ln];
    c = WMMA_F32(a, b, c);
  }
#pragma unroll
  for (int r = 0; r < 8; ++r)
    Q[(size_t)(mt * 16 + hi * 8 + r) * NN + nt * 16 + ln] = c[r];
}

// ---------------------------------------------------------------------------
// Per-chunk local scan: h=0; for t in chunk: h = h@A + u_t   (h is 8x128,
// padded to 16x128 in LDS; rows 8..15 stay zero). Wave w owns N-tile w.
// A fragments for each wave's N-tile are held in 64 VGPRs for the whole loop.
// ---------------------------------------------------------------------------
__global__ void k_chunk_scan(const float* __restrict__ u,
                             const float* __restrict__ A,
                             float* __restrict__ p) {
  __shared__ float hbuf[16 * HSTR];
  int cc = blockIdx.x;                // chunk id
  int l  = threadIdx.x & 31;
  int w  = threadIdx.x >> 5;          // wave -> N tile
  int n0 = w * 16;
  int ln = l & 15;
  int hi = l >> 4;

  for (int i = threadIdx.x; i < 16 * HSTR; i += 256) hbuf[i] = 0.f;

  v2f af[32];
#pragma unroll
  for (int kb = 0; kb < 32; ++kb) {
    int ka = kb * 4 + hi * 2;
    af[kb].x = A[(size_t)ka * NN + n0 + ln];
    af[kb].y = A[(size_t)(ka + 1) * NN + n0 + ln];
  }
  __syncthreads();

  int t0 = cc * CLEN;
  for (int t = 0; t < CLEN; ++t) {
    v8f c;
#pragma unroll
    for (int r = 0; r < 8; ++r)
      c[r] = (hi == 0) ? u[((size_t)r * SS + t0 + t) * NN + n0 + ln] : 0.f;
#pragma unroll
    for (int kb = 0; kb < 32; ++kb) {
      int ka = kb * 4 + hi * 2;
      v2f a;
      a.x = hbuf[ln * HSTR + ka];
      a.y = hbuf[ln * HSTR + ka + 1];
      c = WMMA_F32(a, af[kb], c);
    }
    __syncthreads();
    if (hi == 0) {
#pragma unroll
      for (int r = 0; r < 8; ++r) hbuf[r * HSTR + n0 + ln] = c[r];
    }
    __syncthreads();
  }
  for (int i = threadIdx.x; i < BB * NN; i += 256)
    p[(size_t)cc * BB * NN + i] = hbuf[(i >> 7) * HSTR + (i & 127)];
}

// ---------------------------------------------------------------------------
// Combine chunks: h=0; for c: h = h@M + p_c  (M = A^CLEN). Single workgroup.
// ---------------------------------------------------------------------------
__global__ void k_combine(const float* __restrict__ p,
                          const float* __restrict__ M,
                          float* __restrict__ hout) {
  __shared__ float hbuf[16 * HSTR];
  int l  = threadIdx.x & 31;
  int w  = threadIdx.x >> 5;
  int n0 = w * 16;
  int ln = l & 15;
  int hi = l >> 4;

  for (int i = threadIdx.x; i < 16 * HSTR; i += 256) hbuf[i] = 0.f;

  v2f mf[32];
#pragma unroll
  for (int kb = 0; kb < 32; ++kb) {
    int ka = kb * 4 + hi * 2;
    mf[kb].x = M[(size_t)ka * NN + n0 + ln];
    mf[kb].y = M[(size_t)(ka + 1) * NN + n0 + ln];
  }
  __syncthreads();

  for (int cc = 0; cc < CHUNKS; ++cc) {
    v8f c;
#pragma unroll
    for (int r = 0; r < 8; ++r)
      c[r] = (hi == 0) ? p[((size_t)cc * BB + r) * NN + n0 + ln] : 0.f;
#pragma unroll
    for (int kb = 0; kb < 32; ++kb) {
      int ka = kb * 4 + hi * 2;
      v2f a;
      a.x = hbuf[ln * HSTR + ka];
      a.y = hbuf[ln * HSTR + ka + 1];
      c = WMMA_F32(a, mf[kb], c);
    }
    __syncthreads();
    if (hi == 0) {
#pragma unroll
      for (int r = 0; r < 8; ++r) hbuf[r * HSTR + n0 + ln] = c[r];
    }
    __syncthreads();
  }
  for (int i = threadIdx.x; i < BB * NN; i += 256)
    hout[i] = hbuf[(i >> 7) * HSTR + (i & 127)];
}

// ---------------------------------------------------------------------------
// h_proj[b,d] = sum_n h[b,n] * W_imp[d,n]
// ---------------------------------------------------------------------------
__global__ void k_hproj(const float* __restrict__ h,
                        const float* __restrict__ W,
                        float* __restrict__ hp) {
  int tid = blockIdx.x * 256 + threadIdx.x;  // 8192
  int b = tid >> 10, d = tid & 1023;
  const float* hb = h + b * NN;
  const float* wr = W + (size_t)d * NN;
  float s = 0.f;
#pragma unroll 8
  for (int n = 0; n < NN; ++n) s += hb[n] * wr[n];
  hp[tid] = s;
}

// ---------------------------------------------------------------------------
// importance[b,s] = dot(x[b,s,:], h_proj[b,:]) ; one wave per (b,s) row
// ---------------------------------------------------------------------------
__global__ void k_importance(const float* __restrict__ x,
                             const float* __restrict__ hp,
                             float* __restrict__ imp) {
  int l = threadIdx.x & 31;
  int g = blockIdx.x * (blockDim.x >> 5) + (threadIdx.x >> 5);  // 0..32767
  const float* xr = x + (size_t)g * DD;
  const float* hb = hp + (size_t)(g >> 12) * DD;
  float s = 0.f;
#pragma unroll
  for (int j = 0; j < 8; ++j) {
    int d = (j * 32 + l) * 4;
    float4 xv = *(const float4*)(xr + d);
    float4 hv = *(const float4*)(hb + d);
    s += xv.x * hv.x + xv.y * hv.y + xv.z * hv.z + xv.w * hv.w;
  }
  for (int off = 16; off; off >>= 1) s += __shfl_down(s, off);
  if (l == 0) imp[g] = s;
}

// ---------------------------------------------------------------------------
// Row softmax over S=4096 per batch; 8 workgroups.
// ---------------------------------------------------------------------------
__global__ void k_softmax(const float* __restrict__ imp, float* __restrict__ out) {
  __shared__ float red[256];
  int b = blockIdx.x, tid = threadIdx.x;
  const float* row = imp + (size_t)b * SS;
  float v[16];
  float mx = -3.4e38f;
#pragma unroll
  for (int j = 0; j < 16; ++j) { v[j] = row[j * 256 + tid]; mx = fmaxf(mx, v[j]); }
  red[tid] = mx; __syncthreads();
  for (int o = 128; o; o >>= 1) { if (tid < o) red[tid] = fmaxf(red[tid], red[tid + o]); __syncthreads(); }
  mx = red[0]; __syncthreads();
  float s = 0.f;
#pragma unroll
  for (int j = 0; j < 16; ++j) { v[j] = __expf(v[j] - mx); s += v[j]; }
  red[tid] = s; __syncthreads();
  for (int o = 128; o; o >>= 1) { if (tid < o) red[tid] += red[tid + o]; __syncthreads(); }
  float inv = 1.f / red[0];
#pragma unroll
  for (int j = 0; j < 16; ++j) out[(size_t)b * SS + j * 256 + tid] = v[j] * inv;
}

extern "C" void kernel_launch(void* const* d_in, const int* in_sizes, int n_in,
                              void* d_out, int out_size, void* d_ws, size_t ws_size,
                              hipStream_t stream) {
  const float* x  = (const float*)d_in[0];   // [8,4096,1024]
  const float* A  = (const float*)d_in[1];   // [128,128]
  const float* Bm = (const float*)d_in[2];   // [1024,128]
  const float* W  = (const float*)d_in[3];   // [1024,128]

  char* ws = (char*)d_ws;
  float*  u    = (float*)(ws);                    // 16 MB
  __bf16* pk   = (__bf16*)(ws + 16777216);        // 256 KB packed B_mat
  float*  powa = (float*)(ws + 17039360);         // 64 KB
  float*  powb = (float*)(ws + 17104896);         // 64 KB
  float*  p    = (float*)(ws + 17170432);         // 256 KB chunk partials
  float*  hfin = (float*)(ws + 17432576);         // 4 KB
  float*  hp   = (float*)(ws + 17436672);         // 32 KB
  float*  imp  = (float*)(ws + 17469440);         // 128 KB

  k_packB<<<512, 256, 0, stream>>>(Bm, pk);
  k_gemm_u<<<256, 256, 0, stream>>>(x, pk, u);

  // A^64 by repeated squaring: A->A^2->A^4->...->A^64 (result in powb)
  k_matsq<<<8, 256, 0, stream>>>(A,    powa);
  k_matsq<<<8, 256, 0, stream>>>(powa, powb);
  k_matsq<<<8, 256, 0, stream>>>(powb, powa);
  k_matsq<<<8, 256, 0, stream>>>(powa, powb);
  k_matsq<<<8, 256, 0, stream>>>(powb, powa);
  k_matsq<<<8, 256, 0, stream>>>(powa, powb);

  k_chunk_scan<<<CHUNKS, 256, 0, stream>>>(u, A, p);
  k_combine<<<1, 256, 0, stream>>>(p, powb, hfin);
  k_hproj<<<32, 256, 0, stream>>>(hfin, W, hp);
  k_importance<<<4096, 256, 0, stream>>>(x, hp, imp);
  k_softmax<<<8, 256, 0, stream>>>(imp, (float*)d_out);
}
